// BidirectionalEdgeGraphNetwork_19679540150731
// MI455X (gfx1250) — matrix operand order repaired and checked by reading, not compiled, not run on hardware
//
#include <hip/hip_runtime.h>
#include <hip/hip_bf16.h>
#include <math.h>

// ---------------- problem constants ----------------
#define DN 128
#define DE 128
#define HH 8
#define N_NODES 10000
#define N_EDGES 160000
#define C1_IN 512    // 2*DN + 2*DE
#define C1_OUT 384   // DN + 2*DE
#define BN_INV 0.9999950000374997f   // 1/sqrt(1+1e-5)
#define TEMP_INV 0.25f               // 1/sqrt(DEP), DEP=16

// ---------------- WMMA plumbing (gfx1250, wave32) ----------------
typedef __attribute__((ext_vector_type(16))) __bf16 v16bf;
typedef __attribute__((ext_vector_type(8)))  float  v8f;

union Frag16 { v16bf v; uint4 q[2]; };

static __device__ __forceinline__ v8f wmma_bf16(const Frag16& a, const Frag16& b, v8f c) {
  return __builtin_amdgcn_wmma_f32_16x16x32_bf16(false, a.v, false, b.v, (short)0, c, false, false);
}

// pack 8 f32 -> 8 bf16 (RNE) in one uint4 for a single ds_store_b128
static __device__ __forceinline__ uint4 pack_bf16x8(float4 a, float4 b) {
  union { __bf16 h[8]; uint4 u; } r;
  r.h[0] = (__bf16)a.x; r.h[1] = (__bf16)a.y; r.h[2] = (__bf16)a.z; r.h[3] = (__bf16)a.w;
  r.h[4] = (__bf16)b.x; r.h[5] = (__bf16)b.y; r.h[6] = (__bf16)b.z; r.h[7] = (__bf16)b.w;
  return r.u;
}

// B fragment: W is [Nout x K] row-major bf16.  lane n = nb + (lane&15),
// kb = kc + (lane>=16 ? 8 : 0); K(j) = kb + j for j<8, kb + 16 + (j-8) for j>=8.
static __device__ __forceinline__ Frag16 load_b_glb(const __bf16* __restrict__ W,
                                                    int n, int K, int kb) {
  const __bf16* p = W + (size_t)n * K + kb;
  Frag16 f;
  f.q[0] = *(const uint4*)(p);
  f.q[1] = *(const uint4*)(p + 16);
  return f;
}

// ---------------- generic fused GEMM: out = epilogue(A @ W^T + bias) ----------------
// A: [M x K] f32 (K multiple of 32, K <= 512). W: [N x K] bf16.
// Block = 128 thr (4 waves), tile = 32 rows x 128 cols; wave -> 16x64 subtile
// (4 wmma accumulators). Whole A tile staged in LDS once (barrier-free K loop).
// mode 0: out=acc+bias   1: relu(g*((acc+bias)*BN_INV)+beta)
// mode 2: sigmoid(acc+bias)   3: (acc+bias)*mul[m*N+n]
__global__ __launch_bounds__(128) void gemm_bias(
    const float* __restrict__ A, const __bf16* __restrict__ W,
    const float* __restrict__ bias, const float* __restrict__ g,
    const float* __restrict__ beta, const float* __restrict__ mul,
    float* __restrict__ out, int M, int N, int K, int mode)
{
  __shared__ __align__(16) __bf16 As[32 * 520];   // 32 x K (K<=512), stride K+8
  const int tid  = threadIdx.x;
  const int lane = tid & 31;
  const int wave = tid >> 5;
  const int mhalf = wave >> 1, nq = wave & 1;
  const int hi = lane >> 4;
  const int m0  = blockIdx.x * 32;
  const int nb0 = blockIdx.y * 128 + nq * 64;
  const int stride = K + 8;

  { // stage whole 32 x K A tile as bf16, one b128 store per 8 elements
    const int srow = tid >> 2, sseg = tid & 3;
    const int gr = m0 + srow;
    for (int k = sseg * 8; k < K; k += 32) {
      uint4 v;
      if (gr < M) {
        const float* src = A + (size_t)gr * K + k;
        v = pack_bf16x8(*(const float4*)(src), *(const float4*)(src + 4));
      } else {
        v = uint4{0u, 0u, 0u, 0u};
      }
      *(uint4*)&As[srow * stride + k] = v;
    }
  }
  __syncthreads();

  v8f acc[4] = {v8f{}, v8f{}, v8f{}, v8f{}};
  const __bf16* ar = &As[(mhalf * 16 + (lane & 15)) * stride];

  for (int kc = 0; kc < K; kc += 32) {
    Frag16 a;
    a.q[0] = *(const uint4*)(ar + kc + hi * 8);
    a.q[1] = *(const uint4*)(ar + kc + hi * 8 + 16);
    #pragma unroll
    for (int t = 0; t < 4; ++t) {
      Frag16 b = load_b_glb(W, nb0 + t * 16 + (lane & 15), K, kc + hi * 8);
      acc[t] = wmma_bf16(a, b, acc[t]);
    }
  }

  #pragma unroll
  for (int t = 0; t < 4; ++t) {
    int n = nb0 + t * 16 + (lane & 15);
    float bb = bias ? bias[n] : 0.f;
    float gg = g ? g[n] : 1.f;
    float bt = beta ? beta[n] : 0.f;
    #pragma unroll
    for (int r = 0; r < 8; ++r) {
      int m = m0 + mhalf * 16 + r + hi * 8;
      if (m >= M) continue;
      float v = acc[t][r] + bb;
      if (mode == 1)      { v = gg * (v * BN_INV) + bt; v = v > 0.f ? v : 0.f; }
      else if (mode == 2) { v = 1.f / (1.f + __expf(-v)); }
      else if (mode == 3) { v = v * mul[(size_t)m * N + n]; }
      out[(size_t)m * N + n] = v;
    }
  }
}

// ---------------- edge MLP: ecat[E,512] -> relu(bn(@We1^T)) -> @We2^T + be2 ----------------
// Tile 32 edges per block; ecat staged (gathered) in LDS as bf16 once, He kept in LDS.
__global__ __launch_bounds__(128) void edge_mlp(
    const float* __restrict__ x, const float* __restrict__ ef,
    const int* __restrict__ row, const int* __restrict__ col,
    const int* __restrict__ rev,
    const __bf16* __restrict__ W1, const float* __restrict__ b1,
    const float* __restrict__ g1, const float* __restrict__ bt1,
    const __bf16* __restrict__ W2, const float* __restrict__ b2,
    float* __restrict__ outUE)
{
  __shared__ __align__(16) __bf16 As[32 * 520];   // 32 x 512, stride 520
  __shared__ __align__(16) __bf16 He[32 * 392];   // 32 x 384, stride 392
  const int tid = threadIdx.x;
  const int lane = tid & 31, wave = tid >> 5;
  const int mhalf = wave >> 1, nhalf = wave & 1;
  const int hi = lane >> 4;
  const int e0 = blockIdx.x * 32;

  { // stage gathered concat tile: thread -> (row m = tid>>2, 128-col segment q = tid&3)
    int m = tid >> 2, qseg = tid & 3;
    int e = e0 + m;
    const float* src = nullptr;
    if (e < N_EDGES) {
      if      (qseg == 0) src = x  + (size_t)row[e] * DN;
      else if (qseg == 1) src = ef + (size_t)e * DE;
      else if (qseg == 2) { int r = rev[e]; src = (r >= 0) ? ef + (size_t)r * DE : nullptr; }
      else                src = x  + (size_t)col[e] * DN;
    }
    __bf16* dst = &As[m * 520 + qseg * 128];
    if (src) {
      for (int i = 0; i < 128; i += 8) {
        *(uint4*)(dst + i) =
            pack_bf16x8(*(const float4*)(src + i), *(const float4*)(src + i + 4));
      }
    } else {
      for (int i = 0; i < 128; i += 8) *(uint4*)(dst + i) = uint4{0u, 0u, 0u, 0u};
    }
  }
  __syncthreads();

  const __bf16* ar = &As[(mhalf * 16 + (lane & 15)) * 520];
  // layer 1: this wave covers rows mhalf*16..+15, cols nhalf*192..+191
  for (int nt = 0; nt < 12; ++nt) {
    int nb = nhalf * 192 + nt * 16;
    v8f acc = {};
    for (int kc = 0; kc < C1_IN; kc += 32) {
      Frag16 a;
      a.q[0] = *(const uint4*)(ar + kc + hi * 8);
      a.q[1] = *(const uint4*)(ar + kc + hi * 8 + 16);
      Frag16 b = load_b_glb(W1, nb + (lane & 15), C1_IN, kc + hi * 8);
      acc = wmma_bf16(a, b, acc);
    }
    int n = nb + (lane & 15);
    float bb = b1[n], gg = g1[n], bt = bt1[n];
    #pragma unroll
    for (int r = 0; r < 8; ++r) {
      int m = mhalf * 16 + r + hi * 8;
      float v = acc[r] + bb;
      v = gg * (v * BN_INV) + bt;
      v = v > 0.f ? v : 0.f;
      He[m * 392 + n] = (__bf16)v;   // lanes 0-15 / 16-31 hit two contiguous rows
    }
  }
  __syncthreads();

  const __bf16* hr = &He[(mhalf * 16 + (lane & 15)) * 392];
  // layer 2: cols nhalf*64 + nt*16, nt<4 -> 128 cols total
  for (int nt = 0; nt < 4; ++nt) {
    int nb = nhalf * 64 + nt * 16;
    v8f acc = {};
    for (int kc = 0; kc < C1_OUT; kc += 32) {
      Frag16 a;
      a.q[0] = *(const uint4*)(hr + kc + hi * 8);
      a.q[1] = *(const uint4*)(hr + kc + hi * 8 + 16);
      Frag16 b = load_b_glb(W2, nb + (lane & 15), C1_OUT, kc + hi * 8);
      acc = wmma_bf16(a, b, acc);
    }
    int n = nb + (lane & 15);
    float bb = b2[n];
    #pragma unroll
    for (int r = 0; r < 8; ++r) {
      int e = e0 + mhalf * 16 + r + hi * 8;
      if (e < N_EDGES) outUE[(size_t)e * DE + n] = acc[r] + bb;
    }
  }
}

// ---------------- reverse-edge lookup via hash (stable-argsort min-index semantics) ----------------
#define HASH_BITS 19
#define HASH_SIZE (1u << HASH_BITS)
#define HMASK (HASH_SIZE - 1u)
#define EMPTY64 0xFFFFFFFFFFFFFFFFull

static __device__ __forceinline__ unsigned hmix(unsigned k) {
  k ^= k >> 16; k *= 0x85EBCA6Bu; k ^= k >> 13; k *= 0xC2B2AE35u; k ^= k >> 16;
  return k;
}

__global__ void hash_insert(const int* __restrict__ row, const int* __restrict__ col,
                            unsigned long long* __restrict__ table) {
  int e = blockIdx.x * blockDim.x + threadIdx.x;
  if (e >= N_EDGES) return;
  unsigned long long key = (unsigned long long)row[e] * N_NODES + col[e];
  unsigned long long packed = (key << 20) | (unsigned long long)e;  // min packed == min edge id
  unsigned slot = hmix((unsigned)key) & HMASK;
  for (;;) {
    unsigned long long cur = table[slot];
    if (cur == EMPTY64) {
      unsigned long long old = atomicCAS(&table[slot], EMPTY64, packed);
      if (old == EMPTY64) return;
      cur = old;
    }
    if ((cur >> 20) == key) { atomicMin(&table[slot], packed); return; }
    slot = (slot + 1u) & HMASK;
  }
}

__global__ void hash_lookup(const int* __restrict__ row, const int* __restrict__ col,
                            const unsigned long long* __restrict__ table,
                            int* __restrict__ rev) {
  int e = blockIdx.x * blockDim.x + threadIdx.x;
  if (e >= N_EDGES) return;
  unsigned long long rkey = (unsigned long long)col[e] * N_NODES + row[e];
  unsigned slot = hmix((unsigned)rkey) & HMASK;
  int res = -1;
  for (;;) {
    unsigned long long cur = table[slot];
    if (cur == EMPTY64) break;
    if ((cur >> 20) == rkey) { res = (int)(cur & 0xFFFFFull); break; }
    slot = (slot + 1u) & HMASK;
  }
  rev[e] = res;
}

// ---------------- distance-mask MLP (4 -> 32 -> 1, sigmoid) ----------------
__global__ void dist_mask(const float* __restrict__ pos,
                          const int* __restrict__ row, const int* __restrict__ col,
                          const float* __restrict__ Wd1, const float* __restrict__ bd1,
                          const float* __restrict__ gd, const float* __restrict__ betad,
                          const float* __restrict__ Wd2, const float* __restrict__ bd2,
                          float* __restrict__ dmask) {
  int e = blockIdx.x * blockDim.x + threadIdx.x;
  if (e >= N_EDGES) return;
  int r = row[e], c = col[e];
  float dx = pos[r*3+0]-pos[c*3+0], dy = pos[r*3+1]-pos[c*3+1], dz = pos[r*3+2]-pos[c*3+2];
  float df[4] = {dx, dy, dz, sqrtf(dx*dx + dy*dy + dz*dz)};
  float s2 = 0.f;
  #pragma unroll
  for (int j = 0; j < 32; ++j) {
    float s = bd1[j];
    #pragma unroll
    for (int i = 0; i < 4; ++i) s += df[i] * Wd1[j*4+i];
    s = gd[j] * (s * BN_INV) + betad[j];
    s = s > 0.f ? s : 0.f;
    s2 += s * Wd2[j];
  }
  s2 += bd2[0];
  dmask[e] = 1.f / (1.f + __expf(-s2));
}

// ---------------- per-(edge,head) attention + prob out + segment-max agg ----------------
__global__ __launch_bounds__(256) void attention(
    const float* __restrict__ Q, const float* __restrict__ Kf,
    const float* __restrict__ V, const float* __restrict__ dmask,
    const int* __restrict__ row, const int* __restrict__ col,
    const float* __restrict__ Wa1, const float* __restrict__ ba1,
    const float* __restrict__ ga, const float* __restrict__ betaa,
    const float* __restrict__ Wa2, const float* __restrict__ ba2,
    float* __restrict__ prob_out, unsigned* __restrict__ aggEnc)
{
  __shared__ float sW1[1024], sW2[512], sb1[32], sg[32], sbt[32], sb2[16];
  for (int i = threadIdx.x; i < 1024; i += 256) sW1[i] = Wa1[i];
  for (int i = threadIdx.x; i < 512;  i += 256) sW2[i] = Wa2[i];
  if (threadIdx.x < 32) { sb1[threadIdx.x]=ba1[threadIdx.x]; sg[threadIdx.x]=ga[threadIdx.x]; sbt[threadIdx.x]=betaa[threadIdx.x]; }
  if (threadIdx.x < 16) sb2[threadIdx.x] = ba2[threadIdx.x];
  __syncthreads();

  size_t gid = (size_t)blockIdx.x * 256 + threadIdx.x;
  if (gid >= (size_t)N_EDGES * HH) return;
  int e = (int)(gid >> 3), h = (int)(gid & 7);
  int r = row[e], c = col[e];

  float akin[32];
  #pragma unroll
  for (int p = 0; p < 16; ++p) {
    akin[p]      = Q [(size_t)r * 128 + p * 8 + h];
    akin[16 + p] = Kf[(size_t)e * 128 + p * 8 + h];
  }
  float a1[32];
  #pragma unroll
  for (int o = 0; o < 32; ++o) {
    float s = 0.f;
    const float* w = &sW1[o * 32];
    #pragma unroll
    for (int ci = 0; ci < 32; ++ci) s += akin[ci] * w[ci];
    s += sb1[o];
    s = sg[o] * (s * BN_INV) + sbt[o];
    a1[o] = s > 0.f ? s : 0.f;
  }
  float dm = dmask[e];
  float att[16], mx = -3.0e38f;
  #pragma unroll
  for (int j = 0; j < 16; ++j) {
    float s = 0.f;
    const float* w = &sW2[j * 32];
    #pragma unroll
    for (int o = 0; o < 32; ++o) s += a1[o] * w[o];
    s = (s + sb2[j]) * dm;
    att[j] = s;
    mx = s > mx ? s : mx;
  }
  float sum = 0.f;
  #pragma unroll
  for (int j = 0; j < 16; ++j) { att[j] = __expf((att[j] - mx) * TEMP_INV); sum += att[j]; }
  float inv = 1.f / sum;
  #pragma unroll
  for (int j = 0; j < 16; ++j) {
    float p = att[j] * inv;
    prob_out[(size_t)e * 128 + j * 8 + h] = p;
    float w = p * V[(size_t)c * 128 + j * 8 + h];
    unsigned u = __float_as_uint(w);
    unsigned enc = (u & 0x80000000u) ? ~u : (u | 0x80000000u);   // order-preserving
    atomicMax(&aggEnc[(size_t)r * 128 + j * 8 + h], enc);
  }
}

// ---------------- agg decode + build [x | agg] concat ----------------
__global__ void build_xa(const float* __restrict__ x, const unsigned* __restrict__ aggEnc,
                         float* __restrict__ xa) {
  int i = blockIdx.x * blockDim.x + threadIdx.x;
  if (i >= N_NODES * 128) return;
  int n = i >> 7, c = i & 127;
  unsigned enc = aggEnc[i];
  float a = 0.f;
  if (enc != 0x007FFFFFu) {  // 0x007FFFFF == encoded(-inf) == init value
    unsigned u = (enc & 0x80000000u) ? (enc ^ 0x80000000u) : ~enc;
    a = __uint_as_float(u);
    if (!isfinite(a)) a = 0.f;
  }
  xa[(size_t)n * 256 + c]       = x[i];
  xa[(size_t)n * 256 + 128 + c] = a;
}

// ---------------- twin-edge mean aggregation ----------------
__global__ void twin_sum(const float* __restrict__ ue, const int* __restrict__ row,
                         const int* __restrict__ col, float* __restrict__ sumOut,
                         float* __restrict__ sumIn, float* __restrict__ cntOut,
                         float* __restrict__ cntIn) {
  size_t gid = (size_t)blockIdx.x * blockDim.x + threadIdx.x;
  if (gid >= (size_t)N_EDGES * 32) return;
  int e = (int)(gid >> 5), c4 = (int)(gid & 31);
  int r = row[e], c = col[e];
  #pragma unroll
  for (int i = 0; i < 4; ++i) {
    int cc = c4 * 4 + i;
    float v = ue[(size_t)e * 128 + cc];
    atomicAdd(&sumOut[(size_t)r * 128 + cc], v);
    atomicAdd(&sumIn [(size_t)c * 128 + cc], v);
  }
  if (c4 == 0) { atomicAdd(&cntOut[r], 1.f); atomicAdd(&cntIn[c], 1.f); }
}

__global__ void twin_fin(const float* __restrict__ sumOut, const float* __restrict__ sumIn,
                         const float* __restrict__ cntOut, const float* __restrict__ cntIn,
                         float* __restrict__ twin) {
  int i = blockIdx.x * blockDim.x + threadIdx.x;
  if (i >= N_NODES * 128) return;
  int n = i >> 7, c = i & 127;
  float co = cntOut[n]; co = co < 1.f ? 1.f : co;
  float ci = cntIn[n];  ci = ci < 1.f ? 1.f : ci;
  twin[(size_t)n * 256 + c]       = sumOut[i] / co;
  twin[(size_t)n * 256 + 128 + c] = sumIn[i] / ci;
}

// ---------------- utility fills / converts ----------------
__global__ void fill_u32(unsigned* p, unsigned v, int n) {
  int i = blockIdx.x * blockDim.x + threadIdx.x;
  if (i < n) p[i] = v;
}
__global__ void fill_u64(unsigned long long* p, unsigned long long v, int n) {
  int i = blockIdx.x * blockDim.x + threadIdx.x;
  if (i < n) p[i] = v;
}
__global__ void cvt_bf16(const float* __restrict__ s, __bf16* __restrict__ d, int n) {
  int i = blockIdx.x * blockDim.x + threadIdx.x;
  if (i < n) d[i] = (__bf16)s[i];
}

// ---------------- host launch ----------------
extern "C" void kernel_launch(void* const* d_in, const int* in_sizes, int n_in,
                              void* d_out, int out_size, void* d_ws, size_t ws_size,
                              hipStream_t stream) {
  (void)in_sizes; (void)n_in; (void)out_size; (void)ws_size;
  const float* x    = (const float*)d_in[0];
  const float* ef   = (const float*)d_in[1];
  const float* pos  = (const float*)d_in[2];
  const float* Wq   = (const float*)d_in[3];  const float* bq  = (const float*)d_in[4];
  const float* Wk   = (const float*)d_in[5];  const float* bk  = (const float*)d_in[6];
  const float* Wv   = (const float*)d_in[7];  const float* bv  = (const float*)d_in[8];
  const float* Wd1  = (const float*)d_in[9];  const float* bd1 = (const float*)d_in[10];
  const float* gd   = (const float*)d_in[11]; const float* betad=(const float*)d_in[12];
  const float* Wd2  = (const float*)d_in[13]; const float* bd2 = (const float*)d_in[14];
  const float* We1  = (const float*)d_in[15]; const float* be1 = (const float*)d_in[16];
  const float* ge   = (const float*)d_in[17]; const float* betae=(const float*)d_in[18];
  const float* We2  = (const float*)d_in[19]; const float* be2 = (const float*)d_in[20];
  const float* Wea  = (const float*)d_in[21]; const float* bea = (const float*)d_in[22];
  const float* Wn1  = (const float*)d_in[23]; const float* bn1 = (const float*)d_in[24];
  const float* gn   = (const float*)d_in[25]; const float* betan=(const float*)d_in[26];
  const float* Wn2  = (const float*)d_in[27]; const float* bn2 = (const float*)d_in[28];
  const float* Wa1  = (const float*)d_in[29]; const float* ba1 = (const float*)d_in[30];
  const float* ga   = (const float*)d_in[31]; const float* betaa=(const float*)d_in[32];
  const float* Wa2  = (const float*)d_in[33]; const float* ba2 = (const float*)d_in[34];
  const float* Wl1  = (const float*)d_in[35]; const float* bl1 = (const float*)d_in[36];
  const float* gl   = (const float*)d_in[37]; const float* betal=(const float*)d_in[38];
  const float* Wl2  = (const float*)d_in[39]; const float* bl2 = (const float*)d_in[40];
  const int* eidx   = (const int*)d_in[41];
  const int* row = eidx;
  const int* col = eidx + N_EDGES;

  float* out_final = (float*)d_out;                                  // [N,128]
  float* out_ue    = out_final + (size_t)N_NODES * 128;              // [E,128]
  float* out_prob  = out_ue + (size_t)N_EDGES * 128;                 // [E,16,8]

  // ---- workspace layout ----
  char* ws = (char*)d_ws;
  size_t off = 0;
  auto take = [&](size_t bytes) -> void* {
    void* p = ws + off;
    off += (bytes + 255) & ~(size_t)255;
    return p;
  };
  __bf16* W1bf  = (__bf16*)take((size_t)C1_OUT * C1_IN * 2);
  __bf16* W2bf  = (__bf16*)take((size_t)DE * C1_OUT * 2);
  __bf16* Wqbf  = (__bf16*)take((size_t)DN * DN * 2);
  __bf16* Wkbf  = (__bf16*)take((size_t)DE * DE * 2);
  __bf16* Wvbf  = (__bf16*)take((size_t)DN * DN * 2);
  __bf16* Wn1bf = (__bf16*)take((size_t)256 * 256 * 2);
  __bf16* Wn2bf = (__bf16*)take((size_t)128 * 256 * 2);
  __bf16* Weabf = (__bf16*)take((size_t)128 * 256 * 2);
  __bf16* Wl1bf = (__bf16*)take((size_t)128 * 128 * 2);
  __bf16* Wl2bf = (__bf16*)take((size_t)128 * 128 * 2);
  unsigned long long* hashT = (unsigned long long*)take((size_t)HASH_SIZE * 8);
  int*      rev    = (int*)take((size_t)N_EDGES * 4);
  float*    dmaskB = (float*)take((size_t)N_EDGES * 4);
  float*    Qb     = (float*)take((size_t)N_NODES * 128 * 4);
  float*    Vb     = (float*)take((size_t)N_NODES * 128 * 4);
  float*    Kfb    = (float*)take((size_t)N_EDGES * 128 * 4);
  unsigned* aggEnc = (unsigned*)take((size_t)N_NODES * 128 * 4);
  float*    xa     = (float*)take((size_t)N_NODES * 256 * 4);
  float*    hn     = (float*)take((size_t)N_NODES * 256 * 4);
  float*    un     = (float*)take((size_t)N_NODES * 128 * 4);
  float*    sumOut = (float*)take((size_t)N_NODES * 128 * 4);
  float*    sumIn  = (float*)take((size_t)N_NODES * 128 * 4);
  float*    cntOut = (float*)take((size_t)N_NODES * 4);
  float*    cntIn  = (float*)take((size_t)N_NODES * 4);
  float*    twin   = (float*)take((size_t)N_NODES * 256 * 4);
  float*    Eatt   = (float*)take((size_t)N_NODES * 128 * 4);
  float*    hl     = (float*)take((size_t)N_NODES * 128 * 4);

  auto grid1 = [](long long n, int b) { return (int)((n + b - 1) / b); };

  // ---- init ----
  fill_u64<<<grid1(HASH_SIZE, 256), 256, 0, stream>>>(hashT, EMPTY64, HASH_SIZE);
  fill_u32<<<grid1(N_NODES * 128, 256), 256, 0, stream>>>(aggEnc, 0x007FFFFFu, N_NODES * 128);
  fill_u32<<<grid1(N_NODES * 128, 256), 256, 0, stream>>>((unsigned*)sumOut, 0u, N_NODES * 128);
  fill_u32<<<grid1(N_NODES * 128, 256), 256, 0, stream>>>((unsigned*)sumIn, 0u, N_NODES * 128);
  fill_u32<<<grid1(N_NODES, 256), 256, 0, stream>>>((unsigned*)cntOut, 0u, N_NODES);
  fill_u32<<<grid1(N_NODES, 256), 256, 0, stream>>>((unsigned*)cntIn, 0u, N_NODES);

  // ---- weight converts (bf16 for WMMA) ----
  cvt_bf16<<<grid1(C1_OUT * C1_IN, 256), 256, 0, stream>>>(We1, W1bf, C1_OUT * C1_IN);
  cvt_bf16<<<grid1(DE * C1_OUT, 256), 256, 0, stream>>>(We2, W2bf, DE * C1_OUT);
  cvt_bf16<<<grid1(DN * DN, 256), 256, 0, stream>>>(Wq, Wqbf, DN * DN);
  cvt_bf16<<<grid1(DE * DE, 256), 256, 0, stream>>>(Wk, Wkbf, DE * DE);
  cvt_bf16<<<grid1(DN * DN, 256), 256, 0, stream>>>(Wv, Wvbf, DN * DN);
  cvt_bf16<<<grid1(256 * 256, 256), 256, 0, stream>>>(Wn1, Wn1bf, 256 * 256);
  cvt_bf16<<<grid1(128 * 256, 256), 256, 0, stream>>>(Wn2, Wn2bf, 128 * 256);
  cvt_bf16<<<grid1(128 * 256, 256), 256, 0, stream>>>(Wea, Weabf, 128 * 256);
  cvt_bf16<<<grid1(128 * 128, 256), 256, 0, stream>>>(Wl1, Wl1bf, 128 * 128);
  cvt_bf16<<<grid1(128 * 128, 256), 256, 0, stream>>>(Wl2, Wl2bf, 128 * 128);

  // ---- reverse-edge lookup ----
  hash_insert<<<grid1(N_EDGES, 256), 256, 0, stream>>>(row, col, hashT);
  hash_lookup<<<grid1(N_EDGES, 256), 256, 0, stream>>>(row, col, hashT, rev);

  // ---- distance mask ----
  dist_mask<<<grid1(N_EDGES, 256), 256, 0, stream>>>(pos, row, col, Wd1, bd1, gd, betad, Wd2, bd2, dmaskB);

  // ---- node-level projections Q, V (hoisted out of edge loop) and per-edge K ----
  gemm_bias<<<dim3(grid1(N_NODES, 32), 1), 128, 0, stream>>>(
      x, Wqbf, bq, nullptr, nullptr, nullptr, Qb, N_NODES, 128, 128, 0);
  gemm_bias<<<dim3(grid1(N_NODES, 32), 1), 128, 0, stream>>>(
      x, Wvbf, bv, nullptr, nullptr, nullptr, Vb, N_NODES, 128, 128, 0);
  gemm_bias<<<dim3(grid1(N_EDGES, 32), 1), 128, 0, stream>>>(
      ef, Wkbf, bk, nullptr, nullptr, nullptr, Kfb, N_EDGES, 128, 128, 0);

  // ---- big fused edge MLP -> updated_edge (direct to d_out) ----
  edge_mlp<<<grid1(N_EDGES, 32), 128, 0, stream>>>(
      x, ef, row, col, rev, W1bf, be1, ge, betae, W2bf, be2, out_ue);

  // ---- attention: prob out + segment-max agg ----
  attention<<<grid1((long long)N_EDGES * HH, 256), 256, 0, stream>>>(
      Qb, Kfb, Vb, dmaskB, row, col, Wa1, ba1, ga, betaa, Wa2, ba2, out_prob, aggEnc);

  // ---- node update ----
  build_xa<<<grid1(N_NODES * 128, 256), 256, 0, stream>>>(x, aggEnc, xa);
  gemm_bias<<<dim3(grid1(N_NODES, 32), 2), 128, 0, stream>>>(
      xa, Wn1bf, bn1, gn, betan, nullptr, hn, N_NODES, 256, 256, 1);
  gemm_bias<<<dim3(grid1(N_NODES, 32), 1), 128, 0, stream>>>(
      hn, Wn2bf, bn2, nullptr, nullptr, nullptr, un, N_NODES, 128, 256, 0);

  // ---- twin edge attention ----
  twin_sum<<<grid1((long long)N_EDGES * 32, 256), 256, 0, stream>>>(out_ue, row, col, sumOut, sumIn, cntOut, cntIn);
  twin_fin<<<grid1(N_NODES * 128, 256), 256, 0, stream>>>(sumOut, sumIn, cntOut, cntIn, twin);
  gemm_bias<<<dim3(grid1(N_NODES, 32), 1), 128, 0, stream>>>(
      twin, Weabf, bea, nullptr, nullptr, nullptr, Eatt, N_NODES, 128, 256, 2);

  // ---- final node path: relu(bn(un@Wl1)) @ Wl2 * Eatt ----
  gemm_bias<<<dim3(grid1(N_NODES, 32), 1), 128, 0, stream>>>(
      un, Wl1bf, bl1, gl, betal, nullptr, hl, N_NODES, 128, 128, 1);
  gemm_bias<<<dim3(grid1(N_NODES, 32), 1), 128, 0, stream>>>(
      hl, Wl2bf, bl2, nullptr, nullptr, Eatt, out_final, N_NODES, 128, 128, 3);
}
